// Net_26706106647191
// MI455X (gfx1250) — compile-verified, matrix operand order checked by reading, not run on hardware
//
#include <hip/hip_runtime.h>
#include <hip/hip_bf16.h>

#define H     64
#define BT    16      // batch tile per workgroup (WMMA M)
#define TLEN  2048
#define ILEN  32

typedef __attribute__((ext_vector_type(16))) __bf16 v16bf;
typedef __attribute__((ext_vector_type(8)))  float  v8f;

// gfx1250 has native v_tanh_f32; fall back to exp2/rcp if the builtin is absent.
#if __has_builtin(__builtin_amdgcn_tanhf)
#define TANHF(x) __builtin_amdgcn_tanhf(x)
#else
#define TANHF(x) (2.0f * __builtin_amdgcn_rcpf(1.0f + __builtin_amdgcn_exp2f(-2.885390082f * (x))) - 1.0f)
#endif

// sigmoid(a + 2*bh) = 0.5 + 0.5*tanh(0.5*a + bh)  (bias pre-halved, one fma in, one fma out)
__device__ __forceinline__ float sig_from_acc(float a, float bhalf) {
  return fmaf(0.5f, TANHF(fmaf(0.5f, a, bhalf)), 0.5f);
}

// A-fragment (16xK slab) from row-major bf16 LDS buffer [BT][ld].
// Lane l: M = l%16, khalf = 8*(l/16); elems 0..7 <- K=k0+khalf.., elems 8..15 <- K=k0+16+khalf..
__device__ __forceinline__ v16bf load_a_frag_lds(const __bf16* base, int ld, int k0) {
  const int lane  = threadIdx.x & 31;
  const int m     = lane & 15;
  const int khalf = (lane >> 4) << 3;
  const __bf16* p = base + m * ld + k0 + khalf;
  union { uint4 u[2]; v16bf v; } af;
  af.u[0] = *(const uint4*)(p);
  af.u[1] = *(const uint4*)(p + 16);
  return af.v;
}

// B-fragment of W^T: B[k][n] = W[n][k], W row-major [nrows][ldk] f32 in global.
__device__ __forceinline__ v16bf load_b_frag(const float* __restrict__ W, int ldk,
                                             int n0, int k0) {
  const int lane  = threadIdx.x & 31;
  const int khalf = (lane >> 4) << 3;
  const float* row = W + (size_t)(n0 + (lane & 15)) * ldk + k0;
  v16bf r;
#pragma unroll
  for (int e = 0; e < 8; ++e) {
    r[e]     = (__bf16)row[khalf + e];
    r[8 + e] = (__bf16)row[16 + khalf + e];
  }
  return r;
}

#define WMMA_BF16(A, B, C) \
  __builtin_amdgcn_wmma_f32_16x16x32_bf16(false, (A), false, (B), (short)0, (C), false, false)

__global__ __launch_bounds__(128, 1)
void lstm2_fused_kernel(const float* __restrict__ x,
                        const float* __restrict__ Wih0, const float* __restrict__ Whh0,
                        const float* __restrict__ bih0, const float* __restrict__ bhh0,
                        const float* __restrict__ Wih1, const float* __restrict__ Whh1,
                        const float* __restrict__ bih1, const float* __restrict__ bhh1,
                        const float* __restrict__ Wout, const float* __restrict__ bout,
                        float* __restrict__ y)
{
  __shared__ __bf16 xbuf[2][BT][ILEN];   // double-buffered input tile (bf16)
  __shared__ __bf16 hA[2][BT][H];        // double-buffered layer0 hidden state
  __shared__ __bf16 hB[2][BT][H];        // double-buffered layer1 hidden state
  __shared__ float  ypart[4][BT];        // per-wave output-projection partials

  const int tid  = threadIdx.x;
  const int lane = tid & 31;
  const int wave = tid >> 5;            // 0..3
  const int jt   = wave * 16;           // this wave's hidden-column tile [jt, jt+16)
  const int b0   = blockIdx.x * BT;     // global batch base

  const int nCol  = lane & 15;          // column within tile (C/D frag N)
  const int mBase = (lane >> 4) * 8;    // C/D frag row base
  const int jcol  = jt + nCol;          // hidden column owned by this lane

  // ---------------- init ----------------
  {
    __bf16* pa = &hA[0][0][0];
    __bf16* pb = &hB[0][0][0];
    for (int q = tid; q < 2 * BT * H; q += 128) {
      pa[q] = (__bf16)0.0f;
      pb[q] = (__bf16)0.0f;
    }
    // preload x tile for t=0 (16*32 f32, 4 per thread), convert to bf16
    int fi = tid * 4;
    int xb = fi >> 5, xk = fi & 31;
    const float* xp = x + (size_t)(b0 + xb) * TLEN * ILEN + xk;
#pragma unroll
    for (int q = 0; q < 4; ++q) xbuf[0][xb][xk + q] = (__bf16)xp[q];
  }

  // ----- weight fragments pinned in VGPRs: 4 gate tiles per wave -----
  v16bf wih0[4], whh0[4][2], wih1[4][2], whh1[4][2];
#pragma unroll
  for (int g = 0; g < 4; ++g) {
    const int n0g = 64 * g + jt;
    wih0[g] = load_b_frag(Wih0, ILEN, n0g, 0);
#pragma unroll
    for (int kt = 0; kt < 2; ++kt) {
      whh0[g][kt] = load_b_frag(Whh0, H, n0g, kt * 32);
      wih1[g][kt] = load_b_frag(Wih1, H, n0g, kt * 32);
      whh1[g][kt] = load_b_frag(Whh1, H, n0g, kt * 32);
    }
  }
  // biases folded into the elementwise phase; sigmoid-gate biases pre-halved
  float b0s[4], b1s[4];
#pragma unroll
  for (int g = 0; g < 4; ++g) {
    const int n = 64 * g + jcol;
    const float s = (g == 2) ? 1.0f : 0.5f;   // gate 2 (g-gate) uses tanh directly
    b0s[g] = s * (bih0[n] + bhh0[n]);
    b1s[g] = s * (bih1[n] + bhh1[n]);
  }
  const float woutv = Wout[jcol];
  const float boutv = bout[0];

  // cell state lives entirely in VGPRs (one column x 8 rows per lane, per layer)
  float c0f[8], c1f[8];
#pragma unroll
  for (int r = 0; r < 8; ++r) { c0f[r] = 0.0f; c1f[r] = 0.0f; }

  __syncthreads();

#pragma clang loop unroll_count(1)
  for (int t = 0; t < TLEN; ++t) {
    const int cur = t & 1;
    const int prv = cur ^ 1;

    // ---- finalize y[t-1] ----
    if (t > 0 && tid < BT) {
      float yv = boutv + ypart[0][tid] + ypart[1][tid] + ypart[2][tid] + ypart[3][tid];
      yv = fminf(fmaxf(yv, 0.0f), 1.0f);
      y[(size_t)(b0 + tid) * TLEN + (t - 1)] = yv;
    }

    v8f acc1[4];   // layer-1 accumulators: hh part issued in Phase A, ih part in Phase B
#pragma unroll
    for (int g = 0; g < 4; ++g)
#pragma unroll
      for (int e = 0; e < 8; ++e) acc1[g][e] = 0.0f;

    // ---- Phase A: layer-0 gates + layer-1 hh-part (overlaps elementwise) ----
    {
      v16bf ax  = load_a_frag_lds(&xbuf[cur][0][0], ILEN, 0);
      v16bf ah0 = load_a_frag_lds(&hA[prv][0][0], H, 0);
      v16bf ah1 = load_a_frag_lds(&hA[prv][0][0], H, 32);
      v16bf bh0 = load_a_frag_lds(&hB[prv][0][0], H, 0);
      v16bf bh1 = load_a_frag_lds(&hB[prv][0][0], H, 32);

      v8f acc0[4];
#pragma unroll
      for (int g = 0; g < 4; ++g)
#pragma unroll
        for (int e = 0; e < 8; ++e) acc0[g][e] = 0.0f;

      // layer-0: 4 independent chains, zero C on first multiply
#pragma unroll
      for (int g = 0; g < 4; ++g) acc0[g] = WMMA_BF16(ax,  wih0[g],    acc0[g]);
#pragma unroll
      for (int g = 0; g < 4; ++g) acc0[g] = WMMA_BF16(ah0, whh0[g][0], acc0[g]);
#pragma unroll
      for (int g = 0; g < 4; ++g) acc0[g] = WMMA_BF16(ah1, whh0[g][1], acc0[g]);
      // layer-1 recurrent part (depends only on hB[t-1]) — pipelines under elementwise
#pragma unroll
      for (int g = 0; g < 4; ++g) acc1[g] = WMMA_BF16(bh0, whh1[g][0], acc1[g]);
#pragma unroll
      for (int g = 0; g < 4; ++g) acc1[g] = WMMA_BF16(bh1, whh1[g][1], acc1[g]);

      // prefetch x(t+1) into the other buffer slot
      if (t + 1 < TLEN) {
        int fi = tid * 4;
        int xb = fi >> 5, xk = fi & 31;
        const float* xp = x + (size_t)(b0 + xb) * TLEN * ILEN + (size_t)(t + 1) * ILEN + xk;
#pragma unroll
        for (int q = 0; q < 4; ++q) xbuf[prv][xb][xk + q] = (__bf16)xp[q];
      }

      // in-register LSTM update, biases folded in, all activations via v_tanh
#pragma unroll
      for (int r = 0; r < 8; ++r) {
        float gi = sig_from_acc(acc0[0][r], b0s[0]);
        float gf = sig_from_acc(acc0[1][r], b0s[1]);
        float gg = TANHF(acc0[2][r] + b0s[2]);
        float go = sig_from_acc(acc0[3][r], b0s[3]);
        float c  = fmaf(gf, c0f[r], gi * gg);
        c0f[r]   = c;
        hA[cur][mBase + r][jcol] = (__bf16)(go * TANHF(c));
      }
    }
    __syncthreads();

    // ---- Phase B: layer-1 ih-part + in-register update + y partials ----
    {
      v16bf ai0 = load_a_frag_lds(&hA[cur][0][0], H, 0);
      v16bf ai1 = load_a_frag_lds(&hA[cur][0][0], H, 32);
#pragma unroll
      for (int g = 0; g < 4; ++g) acc1[g] = WMMA_BF16(ai0, wih1[g][0], acc1[g]);
#pragma unroll
      for (int g = 0; g < 4; ++g) acc1[g] = WMMA_BF16(ai1, wih1[g][1], acc1[g]);

      float psum[8];
#pragma unroll
      for (int r = 0; r < 8; ++r) {
        float gi = sig_from_acc(acc1[0][r], b1s[0]);
        float gf = sig_from_acc(acc1[1][r], b1s[1]);
        float gg = TANHF(acc1[2][r] + b1s[2]);
        float go = sig_from_acc(acc1[3][r], b1s[3]);
        float c  = fmaf(gf, c1f[r], gi * gg);
        c1f[r]   = c;
        float h  = go * TANHF(c);
        hB[cur][mBase + r][jcol] = (__bf16)h;
        psum[r]  = h * woutv;     // this lane's column contribution to row m's dot
      }
      // reduce over the 16 lanes (columns) sharing each row group (wave32)
#pragma unroll
      for (int r = 0; r < 8; ++r) {
        float p = psum[r];
        p += __shfl_xor(p, 1, 32);
        p += __shfl_xor(p, 2, 32);
        p += __shfl_xor(p, 4, 32);
        p += __shfl_xor(p, 8, 32);
        psum[r] = p;
      }
      if (nCol == 0) {            // lanes 0 and 16 hold row groups mBase..mBase+7
#pragma unroll
        for (int r = 0; r < 8; ++r) ypart[wave][mBase + r] = psum[r];
      }
    }
    __syncthreads();
  }

  // finalize last timestep
  if (tid < BT) {
    float yv = boutv + ypart[0][tid] + ypart[1][tid] + ypart[2][tid] + ypart[3][tid];
    yv = fminf(fmaxf(yv, 0.0f), 1.0f);
    y[(size_t)(b0 + tid) * TLEN + (TLEN - 1)] = yv;
  }
}

extern "C" void kernel_launch(void* const* d_in, const int* in_sizes, int n_in,
                              void* d_out, int out_size, void* d_ws, size_t ws_size,
                              hipStream_t stream) {
  (void)in_sizes; (void)n_in; (void)out_size; (void)d_ws; (void)ws_size;
  const float* x    = (const float*)d_in[0];
  const float* Wih0 = (const float*)d_in[1];
  const float* Whh0 = (const float*)d_in[2];
  const float* bih0 = (const float*)d_in[3];
  const float* bhh0 = (const float*)d_in[4];
  const float* Wih1 = (const float*)d_in[5];
  const float* Whh1 = (const float*)d_in[6];
  const float* bih1 = (const float*)d_in[7];
  const float* bhh1 = (const float*)d_in[8];
  const float* Wout = (const float*)d_in[9];
  const float* bout = (const float*)d_in[10];
  float* yout = (float*)d_out;

  hipLaunchKernelGGL(lstm2_fused_kernel, dim3(256 / BT), dim3(128), 0, stream,
                     x, Wih0, Whh0, bih0, bhh0, Wih1, Whh1, bih1, bhh1, Wout, bout, yout);
}
